// TransformerDecoderCharacterPredictor_29222957482184
// MI455X (gfx1250) — matrix operand order repaired and checked by reading, not compiled, not run on hardware
//
#include <hip/hip_runtime.h>

#define NLAY   12
#define DMODEL 256
#define NHEAD  8
#define DH     32
#define VOCAB  256
#define BATCH  4
#define SEQ    1024
#define NTOK   (BATCH * SEQ) /* 4096 */

typedef unsigned short u16;
typedef unsigned int   u32;
typedef __attribute__((ext_vector_type(4)))  u32    u32x4;
typedef __attribute__((ext_vector_type(16))) __bf16 v16bf;
typedef __attribute__((ext_vector_type(8)))  float  v8f;

struct BPair { u32x4 lo, hi; };

static __device__ __forceinline__ v16bf make_frag(u32x4 lo, u32x4 hi) {
    BPair p; p.lo = lo; p.hi = hi;
    return __builtin_bit_cast(v16bf, p);
}

static __device__ __forceinline__ u16 f2bf(float f) {
    u32 u = __float_as_uint(f);
    u += 0x7fffu + ((u >> 16) & 1u);   // round-to-nearest-even
    return (u16)(u >> 16);
}

static __device__ __forceinline__ v8f wmma_bf16(v16bf a, v16bf b, v8f c) {
    return __builtin_amdgcn_wmma_f32_16x16x32_bf16(
        /*neg_a=*/false, a, /*neg_b=*/false, b,
        /*c_mod=*/(short)0, c, /*reuse_a=*/false, /*reuse_b=*/false);
}

// ---- xor-reductions within 16-lane rows on the VALU (v_permlane16), no DS pipe
#if __has_builtin(__builtin_amdgcn_permlane16)
#define PL16(v, s1, s2) __builtin_amdgcn_permlane16((v), (v), (int)(s1), (int)(s2), false, false)
#elif __has_builtin(__builtin_amdgcn_permlane16_b32)
#define PL16(v, s1, s2) __builtin_amdgcn_permlane16_b32((v), (v), (int)(s1), (int)(s2), false, false)
#endif

static __device__ __forceinline__ float redmax16(float v) {
#ifdef PL16
    v = fmaxf(v, __int_as_float(PL16(__float_as_int(v), 0x67452301u, 0xefcdab89u))); // xor 1
    v = fmaxf(v, __int_as_float(PL16(__float_as_int(v), 0x54761032u, 0xdcfe98bau))); // xor 2
    v = fmaxf(v, __int_as_float(PL16(__float_as_int(v), 0x32107654u, 0xba98fedcu))); // xor 4
    v = fmaxf(v, __int_as_float(PL16(__float_as_int(v), 0xfedcba98u, 0x76543210u))); // xor 8
#else
#pragma unroll
    for (int m = 8; m >= 1; m >>= 1) v = fmaxf(v, __shfl_xor(v, m, 16));
#endif
    return v;
}

static __device__ __forceinline__ float redsum16(float v) {
#ifdef PL16
    v += __int_as_float(PL16(__float_as_int(v), 0x67452301u, 0xefcdab89u));
    v += __int_as_float(PL16(__float_as_int(v), 0x54761032u, 0xdcfe98bau));
    v += __int_as_float(PL16(__float_as_int(v), 0x32107654u, 0xba98fedcu));
    v += __int_as_float(PL16(__float_as_int(v), 0xfedcba98u, 0x76543210u));
#else
#pragma unroll
    for (int m = 8; m >= 1; m >>= 1) v += __shfl_xor(v, m, 16);
#endif
    return v;
}

// ---------------------------------------------------------------- convert
__global__ void cvt_bf16_kernel(const float* __restrict__ src,
                                u16* __restrict__ dst, int n) {
    int i = blockIdx.x * blockDim.x + threadIdx.x;
    if (i < n) dst[i] = f2bf(src[i]);
}

// ---------------------------------------------------------------- embedding + positional encoding
__global__ void embed_kernel(const int* __restrict__ idx,
                             const float* __restrict__ enc,
                             float* __restrict__ h, u16* __restrict__ hb) {
    int gid = blockIdx.x * blockDim.x + threadIdx.x;   // NTOK*DMODEL threads
    int t = gid >> 8;
    int d = gid & 255;
    int l = t & (SEQ - 1);
    float x = enc[idx[t] * DMODEL + d];
    float ang = 3.14159265358979323846f * (float)l / (float)(SEQ - 1);
    float pe = (d < 128) ? cosf(ang * exp2f((float)d))
                         : sinf(ang * exp2f((float)(d - 128)));
    x += pe;
    h[gid]  = x;
    hb[gid] = f2bf(x);
}

// ---------------------------------------------------------------- WMMA GEMM: out = A[M,K] @ W[N,K]^T + bias
// One wave computes a 32x64 tile: 2 M-frags x 4 N-frags, 8 WMMAs per K-step.
// EPI: 0 = write f32; 1 = relu -> f32 + bf16; 2 = scatter to Q/K/Vt bf16 layouts
template <int EPI>
__global__ void gemm_wmma(const u16* __restrict__ A, const u16* __restrict__ W,
                          const float* __restrict__ bias,
                          float* __restrict__ outF, u16* __restrict__ outB,
                          u16* __restrict__ qb, u16* __restrict__ kb,
                          u16* __restrict__ vtb,
                          int M, int N, int K) {
    int wid  = (blockIdx.x * blockDim.x + threadIdx.x) >> 5;
    int lane = threadIdx.x & 31;
    int tilesN = N >> 6;                       // 64-wide N tiles
    int tM = wid / tilesN;
    int tN = wid - tM * tilesN;
    if (tM * 32 >= M) return;

    int half = lane >> 4;
    int ln   = lane & 15;
    int koffA = half * 8;                      // A K-halves per ISA layout
    int koffB = half * 8;                      // B K-halves per ISA layout

    const u32x4* Ar0 = (const u32x4*)(A + (size_t)(tM * 32 + ln) * K);
    const u32x4* Ar1 = (const u32x4*)(A + (size_t)(tM * 32 + 16 + ln) * K);
    const u32x4* Wr0 = (const u32x4*)(W + (size_t)(tN * 64 + ln) * K);
    const u32x4* Wr1 = (const u32x4*)(W + (size_t)(tN * 64 + 16 + ln) * K);
    const u32x4* Wr2 = (const u32x4*)(W + (size_t)(tN * 64 + 32 + ln) * K);
    const u32x4* Wr3 = (const u32x4*)(W + (size_t)(tN * 64 + 48 + ln) * K);

    v8f acc[2][4];
#pragma unroll
    for (int mi = 0; mi < 2; ++mi)
#pragma unroll
        for (int nj = 0; nj < 4; ++nj)
            acc[mi][nj] = v8f{0.f, 0.f, 0.f, 0.f, 0.f, 0.f, 0.f, 0.f};

    for (int k0 = 0; k0 < K; k0 += 32) {
        int ia0 = (k0 + koffA) >> 3, ia1 = (k0 + koffA + 16) >> 3;
        int ib0 = (k0 + koffB) >> 3, ib1 = (k0 + koffB + 16) >> 3;
        v16bf a0 = make_frag(Ar0[ia0], Ar0[ia1]);
        v16bf a1 = make_frag(Ar1[ia0], Ar1[ia1]);
        v16bf b0 = make_frag(Wr0[ib0], Wr0[ib1]);
        v16bf b1 = make_frag(Wr1[ib0], Wr1[ib1]);
        v16bf b2 = make_frag(Wr2[ib0], Wr2[ib1]);
        v16bf b3 = make_frag(Wr3[ib0], Wr3[ib1]);
        acc[0][0] = wmma_bf16(a0, b0, acc[0][0]);
        acc[0][1] = wmma_bf16(a0, b1, acc[0][1]);
        acc[0][2] = wmma_bf16(a0, b2, acc[0][2]);
        acc[0][3] = wmma_bf16(a0, b3, acc[0][3]);
        acc[1][0] = wmma_bf16(a1, b0, acc[1][0]);
        acc[1][1] = wmma_bf16(a1, b1, acc[1][1]);
        acc[1][2] = wmma_bf16(a1, b2, acc[1][2]);
        acc[1][3] = wmma_bf16(a1, b3, acc[1][3]);
    }

#pragma unroll
    for (int mi = 0; mi < 2; ++mi)
#pragma unroll
    for (int nj = 0; nj < 4; ++nj)
#pragma unroll
    for (int r = 0; r < 8; ++r) {
        int row = tM * 32 + mi * 16 + r + half * 8;
        int col = tN * 64 + nj * 16 + ln;
        float v = acc[mi][nj][r] + bias[col];
        if constexpr (EPI == 0) {
            outF[(size_t)row * N + col] = v;
        } else if constexpr (EPI == 1) {
            v = fmaxf(v, 0.f);
            outF[(size_t)row * N + col] = v;
            outB[(size_t)row * N + col] = f2bf(v);
        } else {  // QKV scatter (N == 768)
            int which = col >> 8;          // 0=q 1=k 2=v
            int d  = col & 255;
            int hh = d >> 5;
            int dh = d & 31;
            int b  = row >> 10;
            int l  = row & (SEQ - 1);
            int bh = b * NHEAD + hh;
            u16 bv = f2bf(v);
            if (which == 0)      qb[((size_t)bh * SEQ + l) * DH + dh] = bv;
            else if (which == 1) kb[((size_t)bh * SEQ + l) * DH + dh] = bv;
            else                 vtb[((size_t)bh * DH + dh) * SEQ + l] = bv;
        }
    }
}

// ---------------------------------------------------------------- causal flash attention (1 wave = 16 query rows)
#define AWAVES 4
__global__ void attn_kernel(const u16* __restrict__ qbuf,
                            const u16* __restrict__ kbuf,
                            const u16* __restrict__ vtbuf,
                            u16* __restrict__ obuf) {
    __shared__ __align__(16) u16 pbuf[AWAVES][16][48];

    int wib  = threadIdx.x >> 5;
    int lane = threadIdx.x & 31;
    int wid  = blockIdx.x * AWAVES + wib;          // 0 .. 2047
    int qt   = wid & 63;                           // query tile
    int bh   = wid >> 6;                           // batch*head
    int half = lane >> 4;
    int ln   = lane & 15;
    int qbase = qt * 16;

    const u16* Q  = qbuf  + (size_t)bh * SEQ * DH;
    const u16* Kp = kbuf  + (size_t)bh * SEQ * DH;
    const u16* Vt = vtbuf + (size_t)bh * DH * SEQ;

    // Q fragment (A layout), loaded once
    int koffA = half * 8;
    const u32x4* qrow = (const u32x4*)(Q + (size_t)(qbase + ln) * DH);
    v16bf qf = make_frag(qrow[koffA >> 3], qrow[(koffA + 16) >> 3]);

    float mrow[8], lrow[8];
    v8f o0 = {0.f, 0.f, 0.f, 0.f, 0.f, 0.f, 0.f, 0.f};
    v8f o1 = {0.f, 0.f, 0.f, 0.f, 0.f, 0.f, 0.f, 0.f};
#pragma unroll
    for (int r = 0; r < 8; ++r) { mrow[r] = -1e30f; lrow[r] = 0.f; }

    const float scale = 0.17677669529663687f;      // 1/sqrt(32)
    int jcmax = qt >> 1;
    for (int jc = 0; jc <= jcmax; ++jc) {
        int jbase = jc * 32;
        // K fragments: B column n = key row jbase+n (contiguous dh)
        const u32x4* k0r = (const u32x4*)(Kp + (size_t)(jbase + ln) * DH + half * 8);
        const u32x4* k1r = (const u32x4*)(Kp + (size_t)(jbase + 16 + ln) * DH + half * 8);
        v16bf kf0 = make_frag(k0r[0], k0r[2]);
        v16bf kf1 = make_frag(k1r[0], k1r[2]);
        v8f z = {0.f, 0.f, 0.f, 0.f, 0.f, 0.f, 0.f, 0.f};
        v8f s0 = wmma_bf16(qf, kf0, z);
        v8f s1 = wmma_bf16(qf, kf1, z);

        float p0[8], p1[8];
#pragma unroll
        for (int r = 0; r < 8; ++r) {
            int qg = qbase + r + half * 8;
            float a0 = (jbase + ln      <= qg) ? s0[r] * scale : -1e30f;
            float a1 = (jbase + 16 + ln <= qg) ? s1[r] * scale : -1e30f;
            float tmax = redmax16(fmaxf(a0, a1));
            float mnew = fmaxf(mrow[r], tmax);
            float corr = __expf(mrow[r] - mnew);
            p0[r] = __expf(a0 - mnew);
            p1[r] = __expf(a1 - mnew);
            float ps = redsum16(p0[r] + p1[r]);
            lrow[r] = lrow[r] * corr + ps;
            mrow[r] = mnew;
            o0[r] *= corr;
            o1[r] *= corr;
        }

        // C-layout probs -> LDS -> A-layout fragment (same-wave, in-order DS)
#pragma unroll
        for (int r = 0; r < 8; ++r) {
            int row = r + half * 8;
            pbuf[wib][row][ln]      = f2bf(p0[r]);
            pbuf[wib][row][16 + ln] = f2bf(p1[r]);
        }
        asm volatile("s_wait_dscnt 0x0" ::: "memory");
        const u32x4* prow = (const u32x4*)&pbuf[wib][ln][0];
        v16bf pf = make_frag(prow[koffA >> 3], prow[(koffA + 16) >> 3]);
        asm volatile("" ::: "memory");

        // V^T fragments: B column n = dim row (contiguous keys)
        const u32x4* v0r = (const u32x4*)(Vt + (size_t)ln * SEQ + jbase + half * 8);
        const u32x4* v1r = (const u32x4*)(Vt + (size_t)(16 + ln) * SEQ + jbase + half * 8);
        v16bf vf0 = make_frag(v0r[0], v0r[2]);
        v16bf vf1 = make_frag(v1r[0], v1r[2]);
        o0 = wmma_bf16(pf, vf0, o0);
        o1 = wmma_bf16(pf, vf1, o1);
    }

    int b  = bh >> 3;
    int hh = bh & 7;
#pragma unroll
    for (int r = 0; r < 8; ++r) {
        int rowg = qbase + r + half * 8;
        float inv = 1.f / lrow[r];
        size_t tok = (size_t)b * SEQ + rowg;
        obuf[tok * DMODEL + hh * DH + ln]      = f2bf(o0[r] * inv);
        obuf[tok * DMODEL + hh * DH + 16 + ln] = f2bf(o1[r] * inv);
    }
}

// ---------------------------------------------------------------- fused add + layernorm (1 wave = 1 token)
__global__ void add_ln_kernel(const float* __restrict__ x, const float* __restrict__ y,
                              const float* __restrict__ g, const float* __restrict__ b,
                              float* __restrict__ outF, u16* __restrict__ outB, int ntok) {
    int wib  = threadIdx.x >> 5;
    int lane = threadIdx.x & 31;
    int t = blockIdx.x * (blockDim.x >> 5) + wib;
    if (t >= ntok) return;
    const float* xr = x + (size_t)t * DMODEL;
    const float* yr = y + (size_t)t * DMODEL;
    float v[8], s = 0.f, s2 = 0.f;
#pragma unroll
    for (int i = 0; i < 8; ++i) {
        int d = i * 32 + lane;
        v[i] = xr[d] + yr[d];
        s  += v[i];
        s2 += v[i] * v[i];
    }
#pragma unroll
    for (int m = 16; m >= 1; m >>= 1) {
        s  += __shfl_xor(s,  m, 32);
        s2 += __shfl_xor(s2, m, 32);
    }
    float mean = s * (1.f / DMODEL);
    float var  = s2 * (1.f / DMODEL) - mean * mean;
    float rstd = rsqrtf(var + 1e-5f);
#pragma unroll
    for (int i = 0; i < 8; ++i) {
        int d = i * 32 + lane;
        float o = g[d] * (v[i] - mean) * rstd + b[d];
        if (outF) outF[(size_t)t * DMODEL + d] = o;
        outB[(size_t)t * DMODEL + d] = f2bf(o);
    }
}

// ---------------------------------------------------------------- host orchestration
extern "C" void kernel_launch(void* const* d_in, const int* in_sizes, int n_in,
                              void* d_out, int out_size, void* d_ws, size_t ws_size,
                              hipStream_t stream) {
    (void)in_sizes; (void)n_in; (void)out_size; (void)ws_size;
    const int*   idx  = (const int*)  d_in[0];
    const float* enc  = (const float*)d_in[1];
    const float* Wqkv = (const float*)d_in[2];
    const float* bqkv = (const float*)d_in[3];
    const float* Wo   = (const float*)d_in[4];
    const float* bo   = (const float*)d_in[5];
    const float* g1   = (const float*)d_in[6];
    const float* be1  = (const float*)d_in[7];
    const float* Wa1  = (const float*)d_in[8];
    const float* ba1  = (const float*)d_in[9];
    const float* Wa2  = (const float*)d_in[10];
    const float* ba2  = (const float*)d_in[11];
    const float* g2   = (const float*)d_in[12];
    const float* be2  = (const float*)d_in[13];
    const float* Wr   = (const float*)d_in[14];
    const float* br   = (const float*)d_in[15];

    char* ws = (char*)d_ws;
    size_t off = 0;
    auto alloc = [&](size_t bytes) {
        void* p = ws + off;
        off += (bytes + 255) & ~(size_t)255;
        return p;
    };
    float* h     = (float*)alloc((size_t)NTOK * DMODEL * 4);
    u16*   hbf   = (u16*)  alloc((size_t)NTOK * DMODEL * 2);
    u16*   qbf   = (u16*)  alloc((size_t)NTOK * DMODEL * 2);
    u16*   kbf   = (u16*)  alloc((size_t)NTOK * DMODEL * 2);
    u16*   vtbf  = (u16*)  alloc((size_t)NTOK * DMODEL * 2);
    u16*   obf   = (u16*)  alloc((size_t)NTOK * DMODEL * 2);
    u16*   h1bf  = (u16*)  alloc((size_t)NTOK * DMODEL * 2);
    u16*   abf   = (u16*)  alloc((size_t)NTOK * DMODEL * 2);
    float* tmp   = (float*)alloc((size_t)NTOK * DMODEL * 4);
    float* af32  = (float*)alloc((size_t)NTOK * DMODEL * 4);
    u16* wqkv_bf = (u16*)alloc((size_t)NLAY * 3 * DMODEL * DMODEL * 2);
    u16* wo_bf   = (u16*)alloc((size_t)NLAY * DMODEL * DMODEL * 2);
    u16* wa1_bf  = (u16*)alloc((size_t)NLAY * DMODEL * DMODEL * 2);
    u16* wa2_bf  = (u16*)alloc((size_t)NLAY * DMODEL * DMODEL * 2);
    u16* wr_bf   = (u16*)alloc((size_t)VOCAB * DMODEL * 2);

    auto cvt = [&](const float* s, u16* d, int n) {
        cvt_bf16_kernel<<<(n + 255) / 256, 256, 0, stream>>>(s, d, n);
    };
    cvt(Wqkv, wqkv_bf, NLAY * 3 * DMODEL * DMODEL);
    cvt(Wo,   wo_bf,   NLAY * DMODEL * DMODEL);
    cvt(Wa1,  wa1_bf,  NLAY * DMODEL * DMODEL);
    cvt(Wa2,  wa2_bf,  NLAY * DMODEL * DMODEL);
    cvt(Wr,   wr_bf,   VOCAB * DMODEL);

    embed_kernel<<<(NTOK * DMODEL) / 256, 256, 0, stream>>>(idx, enc, h, hbf);

    // 32x64 output tile per wave, 8 waves per 256-thread block
    const int tilesM32 = NTOK / 32;                           // 128
    const int blkQKV = tilesM32 * (3 * DMODEL / 64) / 8;      // 192
    const int blkSQ  = tilesM32 * (DMODEL / 64) / 8;          // 64
    const int blkAT  = (32 * (SEQ / 16)) / AWAVES;            // 512
    const int blkLN  = NTOK / 8;                              // 512

    for (int l = 0; l < NLAY; ++l) {
        // qkv = h @ Wqkv^T + bqkv  ->  Q/K/V^T bf16 layouts
        gemm_wmma<2><<<blkQKV, 256, 0, stream>>>(
            hbf, wqkv_bf + (size_t)l * 3 * DMODEL * DMODEL, bqkv + l * 3 * DMODEL,
            nullptr, nullptr, qbf, kbf, vtbf, NTOK, 3 * DMODEL, DMODEL);
        // causal attention
        attn_kernel<<<blkAT, 32 * AWAVES, 0, stream>>>(qbf, kbf, vtbf, obf);
        // y = o @ Wo^T + bo
        gemm_wmma<0><<<blkSQ, 256, 0, stream>>>(
            obf, wo_bf + (size_t)l * DMODEL * DMODEL, bo + l * DMODEL,
            tmp, nullptr, nullptr, nullptr, nullptr, NTOK, DMODEL, DMODEL);
        // h1 = LN(h + y)
        add_ln_kernel<<<blkLN, 256, 0, stream>>>(h, tmp, g1 + l * DMODEL, be1 + l * DMODEL,
                                                 nullptr, h1bf, NTOK);
        // a = relu(h1 @ Wa1^T + ba1)
        gemm_wmma<1><<<blkSQ, 256, 0, stream>>>(
            h1bf, wa1_bf + (size_t)l * DMODEL * DMODEL, ba1 + l * DMODEL,
            af32, abf, nullptr, nullptr, nullptr, NTOK, DMODEL, DMODEL);
        // z = a @ Wa2^T + ba2
        gemm_wmma<0><<<blkSQ, 256, 0, stream>>>(
            abf, wa2_bf + (size_t)l * DMODEL * DMODEL, ba2 + l * DMODEL,
            tmp, nullptr, nullptr, nullptr, nullptr, NTOK, DMODEL, DMODEL);
        // h = LN(a + z)
        add_ln_kernel<<<blkLN, 256, 0, stream>>>(af32, tmp, g2 + l * DMODEL, be2 + l * DMODEL,
                                                 h, hbf, NTOK);
    }

    // logits = h @ Wr^T + br
    gemm_wmma<0><<<blkSQ, 256, 0, stream>>>(
        hbf, wr_bf, br, (float*)d_out, nullptr, nullptr, nullptr, nullptr,
        NTOK, VOCAB, DMODEL);
}
